// SpectralTemporalProcessor_78211354460520
// MI455X (gfx1250) — matrix operand order repaired and checked by reading, not compiled, not run on hardware
//
#include <hip/hip_runtime.h>
#include <hip/hip_bf16.h>

// ---------------------------------------------------------------------------
// Continuous wavelet transform as Toeplitz-matmul on CDNA5 WMMA (gfx1250).
//   y[b,i,s,f] = sw[i] * sum_l wdil_i[l] * x[b, s-l, f]
// Pass 1a: signal fp32 (B,S,F) -> fp16 transposed (B,F,S) in ws (L2-resident).
// Pass 1b: reversed/padded dilated f16 wavelets -> ws (computed once).
// Pass 2 : per wave: one feature block x two s-tiles. Per K-block:
//          1x B load (2 global_load_b128, shared by both tiles),
//          2x A load (merged ds_load_b128 at +/-32B offsets),
//          2x independent v_wmma_f32_16x16x32_f16. No inner barriers.
// ---------------------------------------------------------------------------

typedef __attribute__((ext_vector_type(16))) _Float16 v16h;
typedef __attribute__((ext_vector_type(8)))  _Float16 v8h;
typedef __attribute__((ext_vector_type(8)))  float    v8f;

#define N_SCALES 8
#define WLEN     64
#define BATCH    16
#define SEQ      4096
#define FEAT     64
#define WPPAD    64              // zero pad in front of reversed wavelet
#define WPMAX    (4096 + 128)    // padded reversed wavelet capacity (halfs)
#define XROW     40              // fallback: LDS row stride (halfs)
#define WSIMG    ((size_t)BATCH * FEAT * SEQ)   // f16 image elems in ws

// L_i = int(64 * 64^(i/7)), scales = 64^(i/7); computed offline in fp64.
__constant__ int   kDilLen[N_SCALES] = {64, 115, 210, 380, 689, 1248, 2261, 4096};
__constant__ float kScale [N_SCALES] = {1.0f, 1.81144733f, 3.28134143f, 5.94397776f,
                                        10.76720158f, 19.50421900f, 35.33090500f, 64.0f};

// ---------------------------------------------------------------------------
// Pass 1a: fp32 (B,S,F) -> fp16 (B,F,S) via LDS transpose. 64-seq tiles.
// ---------------------------------------------------------------------------
__global__ __launch_bounds__(256)
void sig_to_f16_transpose(const float* __restrict__ sig, _Float16* __restrict__ wsT)
{
    __shared__ __align__(16) _Float16 XT[FEAT][72];
    const int tid = threadIdx.x;
    const int b   = blockIdx.x >> 6;
    const int s0  = (blockIdx.x & 63) * 64;
    const float* sigB = sig + (size_t)b * SEQ * FEAT;

    #pragma unroll
    for (int r = 0; r < 4; ++r) {
        const int e  = tid + r * 256;
        const int kr = e >> 4;
        const int fq = e & 15;
        const float4 vx = *(const float4*)(sigB + (size_t)(s0 + kr) * FEAT + fq * 4);
        XT[fq * 4 + 0][kr] = (_Float16)vx.x;
        XT[fq * 4 + 1][kr] = (_Float16)vx.y;
        XT[fq * 4 + 2][kr] = (_Float16)vx.z;
        XT[fq * 4 + 3][kr] = (_Float16)vx.w;
    }
    __syncthreads();
    #pragma unroll
    for (int r = 0; r < 2; ++r) {
        const int e  = tid + r * 256;
        const int fr = e >> 3;
        const int sg = e & 7;
        *(v8h*)(wsT + ((size_t)b * FEAT + fr) * SEQ + s0 + sg * 8) =
            *(const v8h*)&XT[fr][sg * 8];
    }
}

// ---------------------------------------------------------------------------
// Pass 1b: one workgroup per scale builds the reversed, zero-padded, dilated
// wavelet (scale_weight and 1/sqrt(scale) folded) as f16 into ws.
// Layout: wsWav[isc*WPMAX + WPPAD + u] = w_isc[L-1-u].
// ---------------------------------------------------------------------------
__global__ __launch_bounds__(256)
void build_wavelets_ws(const float* __restrict__ mw, const float* __restrict__ sw,
                       _Float16* __restrict__ wsWav)
{
    const int isc = blockIdx.x;
    const int tid = threadIdx.x;
    const int   L     = kDilLen[isc];
    const float scale = kScale[isc];
    _Float16* dst = wsWav + (size_t)isc * WPMAX;

    for (int k = tid; k < WPMAX; k += 256) dst[k] = (_Float16)0.0f;
    __syncthreads();

    const float norm = sw[isc] / sqrtf(scale);
    const float step = (L > 1) ? (float)(WLEN - 1) / (float)(L - 1) : 0.0f;
    const float* mrow = mw + isc * WLEN;
    for (int t = tid; t < L; t += 256) {
        float pos = t * step;
        int   i0  = (int)pos;
        if (i0 > WLEN - 2) i0 = WLEN - 2;
        float fr = pos - (float)i0;
        float v  = mrow[i0] * (1.0f - fr) + mrow[i0 + 1] * fr;  // align_corners interp
        dst[WPPAD + (L - 1 - t)] = (_Float16)(v * norm);
    }
}

// ---------------------------------------------------------------------------
// Pass 2: main WMMA kernel. Workgroup = (scale, batch, 64-seq block); 8 waves =
// 4 feature blocks x 2 s-tile pairs; each wave computes two 16x16 tiles.
// ---------------------------------------------------------------------------
__global__ __launch_bounds__(256)
void cwt_wmma_l2(const _Float16* __restrict__ wsT,
                 const _Float16* __restrict__ wsWav,
                 float* __restrict__ out)
{
    __shared__ __align__(16) _Float16 WP[WPMAX];

    const int tid = threadIdx.x;
    const int blk = blockIdx.x;
    const int sb  = blk & 63;                 // 64-seq block
    const int b   = (blk >> 6) & (BATCH - 1);
    const int isc = blk >> 10;

    const int L   = kDilLen[isc];
    const int s0b = sb * 64;

    // Copy precomputed wavelet into LDS (vectorized, one barrier).
    {
        const _Float16* src = wsWav + (size_t)isc * WPMAX;
        for (int c = tid; c < WPMAX / 8; c += 256)
            *(v8h*)&WP[c * 8] = *(const v8h*)(src + c * 8);
    }
    __syncthreads();

    // Wave -> tiles: 4 feature blocks x 2 s-pairs; tiles at s0a and s0a+16.
    const int lane = tid & 31;
    const int wv   = tid >> 5;
    const int s0a  = s0b + 32 * (wv & 1);
    const int n0   = (wv >> 1) * 16;
    const int mro  = lane & 15;          // A row
    const int hgrp = lane >> 4;          // half-wave K group
    const int koff = hgrp * 8;           // B K offset
    const int f    = n0 + (lane & 15);   // B column (feature)

    const _Float16* wsRow = wsT + ((size_t)b * FEAT + f) * SEQ;

    int lo = s0a - L + 1; if (lo < 0) lo = 0;
    const int xb0 = lo & ~31;
    const int nIt = ((s0a - xb0) >> 5) + 1;   // last K-block base is s0a

    v8f accA = {}, accB = {};

    for (int it = 0; it < nIt; ++it) {
        const int xb = xb0 + (it << 5);

        // A fragments for both s-tiles: A[m][k] = w[(s0t-xb)+m-k]; tile1 is
        // tile0 shifted by 16 in k -> same LDS base, offsets differ by 32B.
        const int ub0 = WPPAD + (L - 1) - (s0a - xb) - mro;   // s-tile 0
        const int ub1 = ub0 - 16;                              // s-tile 1 (s0a+16)
        v16h a0, a1;
        #pragma unroll
        for (int j = 0; j < 8; ++j) {
            a0[j]     = WP[ub0 + 8 * hgrp + j];
            a0[j + 8] = WP[ub0 + 16 + 8 * hgrp + j];
            a1[j]     = WP[ub1 + 8 * hgrp + j];
            a1[j + 8] = WP[ub1 + 16 + 8 * hgrp + j];
        }

        // Single shared B fragment: two aligned 16B loads from L2-resident row.
        const v8h blo = *(const v8h*)(wsRow + xb + koff);
        const v8h bhi = *(const v8h*)(wsRow + xb + koff + 16);
        v16h bfrag;
        #pragma unroll
        for (int j = 0; j < 8; ++j) { bfrag[j] = blo[j]; bfrag[j + 8] = bhi[j]; }

        // WGP-scope speculative prefetch of this lane's stream, 2KB ahead.
        asm volatile("global_prefetch_b8 %0, off"
                     :: "v"((unsigned long long)(size_t)(wsRow + xb + 1024))
                     : "memory");

        // Two independent accumulation chains (breaks WMMA RAW hazard).
        accA = __builtin_amdgcn_wmma_f32_16x16x32_f16(
                   false, a0, false, bfrag, (short)0, accA, false, false);
        accB = __builtin_amdgcn_wmma_f32_16x16x32_f16(
                   false, a1, false, bfrag, (short)0, accB, false, false);
    }

    float* outB = out + ((size_t)b * N_SCALES + isc) * (size_t)SEQ * FEAT;
    const int ncol = n0 + (lane & 15);
    #pragma unroll
    for (int r = 0; r < 8; ++r) {
        const int m = r + 8 * hgrp;
        outB[(size_t)(s0a + m) * FEAT + ncol]      = accA[r];
        outB[(size_t)(s0a + 16 + m) * FEAT + ncol] = accB[r];
    }
}

// ---------------------------------------------------------------------------
// Fallback (ws too small): self-contained, per-iteration LDS staging.
// ---------------------------------------------------------------------------
__global__ __launch_bounds__(256)
void cwt_wmma_fallback(const float* __restrict__ sig,
                       const float* __restrict__ mw,
                       const float* __restrict__ sw,
                       float* __restrict__ out)
{
    __shared__ __align__(16) _Float16 WP[WPMAX];
    __shared__ __align__(16) _Float16 XS[FEAT][XROW];

    const int tid  = threadIdx.x;
    const int blk  = blockIdx.x;
    const int sblk = blk & 127;
    const int b    = (blk >> 7) & (BATCH - 1);
    const int isc  = blk >> 11;

    const int   L     = kDilLen[isc];
    const float scale = kScale[isc];
    const int   s0b   = sblk * 32;

    for (int k = tid; k < WPMAX; k += 256) WP[k] = (_Float16)0.0f;
    __syncthreads();
    {
        const float norm = sw[isc] / sqrtf(scale);
        const float step = (L > 1) ? (float)(WLEN - 1) / (float)(L - 1) : 0.0f;
        const float* mrow = mw + isc * WLEN;
        for (int t = tid; t < L; t += 256) {
            float pos = t * step;
            int   i0  = (int)pos;
            if (i0 > WLEN - 2) i0 = WLEN - 2;
            float fr = pos - (float)i0;
            float v  = mrow[i0] * (1.0f - fr) + mrow[i0 + 1] * fr;
            WP[WPPAD + (L - 1 - t)] = (_Float16)(v * norm);
        }
    }

    const int lane = tid & 31;
    const int wv   = tid >> 5;
    const int s0t  = s0b + 16 * (wv & 1);
    const int n0   = (wv >> 1) * 16;
    const int mro  = lane & 15;
    const int hgrp = lane >> 4;

    const float* sigB = sig + (size_t)b * SEQ * FEAT;

    int lo = s0b - L + 1; if (lo < 0) lo = 0;
    const int xb0   = lo & ~31;
    const int nIter = ((s0b - xb0) >> 5) + 1;

    v8f acc = {};

    for (int it = 0; it < nIter; ++it) {
        const int xb = xb0 + (it << 5);
        __syncthreads();
        #pragma unroll
        for (int r = 0; r < 2; ++r) {
            const int e  = tid + r * 256;
            const int kr = e >> 4;
            const int fq = e & 15;
            const float4 vx = *(const float4*)(sigB + (size_t)(xb + kr) * FEAT + fq * 4);
            XS[fq * 4 + 0][kr] = (_Float16)vx.x;
            XS[fq * 4 + 1][kr] = (_Float16)vx.y;
            XS[fq * 4 + 2][kr] = (_Float16)vx.z;
            XS[fq * 4 + 3][kr] = (_Float16)vx.w;
        }
        __syncthreads();

        const int ubase = WPPAD + (L - 1) - (s0t - xb) - mro;
        v16h afrag;
        #pragma unroll
        for (int j = 0; j < 8; ++j) {
            afrag[j]     = WP[ubase + 8 * hgrp + j];
            afrag[j + 8] = WP[ubase + 16 + 8 * hgrp + j];
        }
        const int fi   = n0 + (lane & 15);
        const int koff = hgrp * 8;
        const v8h blo = *(const v8h*)&XS[fi][koff];
        const v8h bhi = *(const v8h*)&XS[fi][koff + 16];
        v16h bfrag;
        #pragma unroll
        for (int j = 0; j < 8; ++j) { bfrag[j] = blo[j]; bfrag[j + 8] = bhi[j]; }

        acc = __builtin_amdgcn_wmma_f32_16x16x32_f16(
                  false, afrag, false, bfrag, (short)0, acc, false, false);
    }

    float* outB = out + ((size_t)b * N_SCALES + isc) * (size_t)SEQ * FEAT;
    const int ncol = n0 + (lane & 15);
    #pragma unroll
    for (int r = 0; r < 8; ++r) {
        const int m = r + 8 * hgrp;
        outB[(size_t)(s0t + m) * FEAT + ncol] = acc[r];
    }
}

extern "C" void kernel_launch(void* const* d_in, const int* in_sizes, int n_in,
                              void* d_out, int out_size, void* d_ws, size_t ws_size,
                              hipStream_t stream) {
    (void)in_sizes; (void)n_in; (void)out_size;
    const float* sig = (const float*)d_in[0];   // (16, 4096, 64) fp32
    const float* mw  = (const float*)d_in[1];   // (8, 64) fp32
    const float* sw  = (const float*)d_in[2];   // (8,) fp32
    float* out = (float*)d_out;                 // (16, 8, 4096, 64) fp32

    const size_t need = (WSIMG + (size_t)N_SCALES * WPMAX) * sizeof(_Float16);

    if (ws_size >= need) {
        _Float16* wsT   = (_Float16*)d_ws;          // f16 (B,F,S) image
        _Float16* wsWav = wsT + WSIMG;              // f16 padded wavelets
        sig_to_f16_transpose<<<BATCH * (SEQ / 64), 256, 0, stream>>>(sig, wsT);
        build_wavelets_ws<<<N_SCALES, 256, 0, stream>>>(mw, sw, wsWav);
        cwt_wmma_l2<<<N_SCALES * BATCH * 64, 256, 0, stream>>>(wsT, wsWav, out);
    } else {
        cwt_wmma_fallback<<<N_SCALES * BATCH * 128, 256, 0, stream>>>(sig, mw, sw, out);
    }
}
